// DCRNN_41626823033599
// MI455X (gfx1250) — matrix operand order repaired and checked by reading, not compiled, not run on hardware
//
#include <hip/hip_runtime.h>

#define NN 50000
#define EE 800000
#define DD 64
#define G3D 192   // 3*D gate rows per weight matrix

typedef __attribute__((ext_vector_type(16))) _Float16 v16h;
typedef __attribute__((ext_vector_type(8)))  _Float16 v8h;
typedef __attribute__((ext_vector_type(8)))  float    v8f;

// ---- fast transcendentals (hardware v_rcp_f32 / v_exp_f32 / v_tanh_f32) ----
static __device__ __forceinline__ float fast_rcp(float x) {
#if __has_builtin(__builtin_amdgcn_rcpf)
  return __builtin_amdgcn_rcpf(x);
#else
  return 1.f / x;
#endif
}
static __device__ __forceinline__ float fast_sigmoid(float x) {
  return fast_rcp(1.f + __expf(-x));
}
static __device__ __forceinline__ float fast_tanh(float x) {
#if __has_builtin(__builtin_amdgcn_tanhf)
  return __builtin_amdgcn_tanhf(x);          // gfx1250 v_tanh_f32 trans op
#else
  return 1.f - 2.f * fast_rcp(__expf(2.f * x) + 1.f);
#endif
}

// D = A(16x32 f16) * B(32x16 f16) + C(16x16 f32)  -- CDNA5 wave32 WMMA
static __device__ __forceinline__ v8f wmma32(v16h a, v16h b, v8f c) {
  return __builtin_amdgcn_wmma_f32_16x16x32_f16(
      /*neg_a=*/false, a, /*neg_b=*/false, b,
      /*c_mod=*/(short)0, c, /*reuse_a=*/false, /*reuse_b=*/false);
}

// A-matrix (16x32, 16-bit) fragment, contiguous k order used by this code (same as B):
//  lane L (M = L&15, hi = L>=16): half i holds K = k0 + 16*hi + i
static __device__ __forceinline__ v16h load_A16(const float* __restrict__ row,
                                                int k0, int hi) {
  const float* p1 = row + k0 + hi * 16;
  const float* p2 = row + k0 + hi * 16 + 8;
  float4 a0 = *(const float4*)(p1);
  float4 a1 = *(const float4*)(p1 + 4);
  float4 b0 = *(const float4*)(p2);
  float4 b1 = *(const float4*)(p2 + 4);
  v16h r;
  r[0]  = (_Float16)a0.x; r[1]  = (_Float16)a0.y; r[2]  = (_Float16)a0.z; r[3]  = (_Float16)a0.w;
  r[4]  = (_Float16)a1.x; r[5]  = (_Float16)a1.y; r[6]  = (_Float16)a1.z; r[7]  = (_Float16)a1.w;
  r[8]  = (_Float16)b0.x; r[9]  = (_Float16)b0.y; r[10] = (_Float16)b0.z; r[11] = (_Float16)b0.w;
  r[12] = (_Float16)b1.x; r[13] = (_Float16)b1.y; r[14] = (_Float16)b1.z; r[15] = (_Float16)b1.w;
  return r;
}

// B-matrix (32x16, 16-bit) fragment, contiguous k order used by this code (same as A):
//  lane L (N = L&15, hi = L>=16): half i holds K = k0 + 16*hi + i
//  B[k][n] = W[n][k]; W pre-converted to f16, row-major [192,64].
//  Per-lane fragment = 16 contiguous halves (32 B, 32 B aligned) -> 2x b128 loads.
static __device__ __forceinline__ v16h load_B16h(const _Float16* __restrict__ W,
                                                 int n, int k0, int hi) {
  return *(const v16h*)(W + (size_t)n * DD + k0 + hi * 16);
}

__global__ void zero_kernel(float4* __restrict__ p, int n4) {
  int i = blockIdx.x * blockDim.x + threadIdx.x;
  if (i < n4) p[i] = make_float4(0.f, 0.f, 0.f, 0.f);
}

// One-time f32 -> f16 weight conversion (n multiple of 8).
__global__ void cvt_f16_kernel(const float* __restrict__ src,
                               _Float16* __restrict__ dst, int n) {
  int base = (blockIdx.x * blockDim.x + threadIdx.x) * 8;
  if (base >= n) return;
  float4 a = *(const float4*)(src + base);
  float4 b = *(const float4*)(src + base + 4);
  v8h o;
  o[0] = (_Float16)a.x; o[1] = (_Float16)a.y; o[2] = (_Float16)a.z; o[3] = (_Float16)a.w;
  o[4] = (_Float16)b.x; o[5] = (_Float16)b.y; o[6] = (_Float16)b.z; o[7] = (_Float16)b.w;
  *(v8h*)(dst + base) = o;
}

// 16 lanes per edge; each lane: one float4 gather (L2-resident) + 4 f32 atomic adds.
__global__ __launch_bounds__(256) void scatter_kernel(
    const float* __restrict__ x, const long long* __restrict__ ei,
    const float* __restrict__ ew, float* __restrict__ aggr) {
  int t = blockIdx.x * blockDim.x + threadIdx.x;
  int e = t >> 4;
  if (e >= EE) return;
  int q = t & 15;
  int src = (int)ei[e];        // edge_index[0, e]
  int dst = (int)ei[EE + e];   // edge_index[1, e]
  float w = ew[e];
  float4 v = *(const float4*)(x + (size_t)src * DD + q * 4);
  float* a = aggr + (size_t)dst * DD + q * 4;
  atomicAdd(a + 0, v.x * w);
  atomicAdd(a + 1, v.y * w);
  atomicAdd(a + 2, v.z * w);
  atomicAdd(a + 3, v.w * w);
}

// One wave32 per 16-node tile.
//   r = sigmoid(i_r + h_r):  single WMMA accumulator chained over 4 matmul-tiles
//   z = sigmoid(i_z + h_z):  same
//   n = tanh(i_n + r*h_n):   i_n and h_n kept separate (r gates h_n only)
//   out = n + z*(h - n)
// All 12 B fragments of an f-iteration are loaded into distinct registers
// BEFORE the WMMA chain so the loads form one clause and latency overlaps
// with the matrix pipe (partial s_wait_loadcnt instead of wait-0 per WMMA).
__global__ __launch_bounds__(256) void gru_kernel(
    const float* __restrict__ aggr, const float* __restrict__ hin,
    const _Float16* __restrict__ wih, const _Float16* __restrict__ whh,
    const float* __restrict__ bih, const float* __restrict__ bhh,
    float* __restrict__ out) {
  const int tile = blockIdx.x * 8 + (threadIdx.x >> 5);
  if (tile >= NN / 16) return;               // wave-uniform guard (EXEC all-1s below)
  const int lane = threadIdx.x & 31;
  const int m = lane & 15;
  const int hi = lane >> 4;
  const int rowBase = tile * 16;

  __builtin_prefetch(wih, 0, 2);             // global_prefetch_b8: warm f16 weights
  __builtin_prefetch(whh, 0, 2);

  const float* rowA = aggr + (size_t)(rowBase + m) * DD;
  const float* rowH = hin + (size_t)(rowBase + m) * DD;
  const v16h aA0 = load_A16(rowA, 0, hi);
  const v16h aA1 = load_A16(rowA, 32, hi);
  const v16h aH0 = load_A16(rowH, 0, hi);
  const v16h aH1 = load_A16(rowH, 32, hi);

#pragma unroll
  for (int f = 0; f < 4; ++f) {
    const int cR = f * 16 + m;          // r-gate column for this lane
    const int cZ = 64 + f * 16 + m;     // z-gate column
    const int cN = 128 + f * 16 + m;    // n-gate column

    // ---- bulk-load all 12 B fragments (24x global_load_b128, one clause) ----
    v16h bR0 = load_B16h(wih, cR, 0, hi);
    v16h bR1 = load_B16h(wih, cR, 32, hi);
    v16h bR2 = load_B16h(whh, cR, 0, hi);
    v16h bR3 = load_B16h(whh, cR, 32, hi);
    v16h bZ0 = load_B16h(wih, cZ, 0, hi);
    v16h bZ1 = load_B16h(wih, cZ, 32, hi);
    v16h bZ2 = load_B16h(whh, cZ, 0, hi);
    v16h bZ3 = load_B16h(whh, cZ, 32, hi);
    v16h bN0 = load_B16h(wih, cN, 0, hi);
    v16h bN1 = load_B16h(wih, cN, 32, hi);
    v16h bN2 = load_B16h(whh, cN, 0, hi);
    v16h bN3 = load_B16h(whh, cN, 32, hi);

    // ---- WMMA chains ----
    v8f accR = {};
    accR = wmma32(aA0, bR0, accR);
    accR = wmma32(aA1, bR1, accR);
    accR = wmma32(aH0, bR2, accR);
    accR = wmma32(aH1, bR3, accR);
    v8f accZ = {};
    accZ = wmma32(aA0, bZ0, accZ);
    accZ = wmma32(aA1, bZ1, accZ);
    accZ = wmma32(aH0, bZ2, accZ);
    accZ = wmma32(aH1, bZ3, accZ);
    v8f iN = {}; v8f hN = {};
    iN = wmma32(aA0, bN0, iN);
    iN = wmma32(aA1, bN1, iN);
    hN = wmma32(aH0, bN2, hN);
    hN = wmma32(aH1, bN3, hN);

    const float bR  = bih[cR] + bhh[cR];
    const float bZ  = bih[cZ] + bhh[cZ];
    const float biN = bih[cN];
    const float bhN = bhh[cN];

#pragma unroll
    for (int v = 0; v < 8; ++v) {
      const int row = rowBase + v + hi * 8;  // C/D layout: M = v + 8*hi
      const int col = f * 16 + m;            // feature index for this lane
      float r = fast_sigmoid(accR[v] + bR);
      float z = fast_sigmoid(accZ[v] + bZ);
      float n = fast_tanh(iN[v] + biN + r * (hN[v] + bhN));
      float hp = hin[(size_t)row * DD + col];
      out[(size_t)row * DD + col] = n + z * (hp - n);   // (1-z)*n + z*hp
    }
  }
}

extern "C" void kernel_launch(void* const* d_in, const int* in_sizes, int n_in,
                              void* d_out, int out_size, void* d_ws, size_t ws_size,
                              hipStream_t stream) {
  const float* x        = (const float*)d_in[0];
  const long long* ei   = (const long long*)d_in[1];  // jnp.int64 in reference
  const float* ew       = (const float*)d_in[2];
  const float* wih1     = (const float*)d_in[3];
  const float* whh1     = (const float*)d_in[4];
  const float* bih1     = (const float*)d_in[5];
  const float* bhh1     = (const float*)d_in[6];
  const float* wih2     = (const float*)d_in[7];
  const float* whh2     = (const float*)d_in[8];
  const float* bih2     = (const float*)d_in[9];
  const float* bhh2     = (const float*)d_in[10];
  float* out  = (float*)d_out;

  // Workspace layout (all offsets 256B-aligned):
  float* aggr = (float*)d_ws;                              // [N,64]  f32
  float* h1   = aggr + (size_t)NN * DD;                    // [N,64]  f32
  _Float16* w16 = (_Float16*)(h1 + (size_t)NN * DD);       // 4 x [192,64] f16
  _Float16* wih1h = w16;
  _Float16* whh1h = w16 + (size_t)1 * G3D * DD;
  _Float16* wih2h = w16 + (size_t)2 * G3D * DD;
  _Float16* whh2h = w16 + (size_t)3 * G3D * DD;

  const int nW  = G3D * DD;                     // 12288 elements per matrix
  const int cvtBlocks  = (nW / 8 + 255) / 256;  // 8 elems per thread
  const int n4 = NN * DD / 4;
  const int zeroBlocks = (n4 + 255) / 256;
  const int scatBlocks = (EE * 16) / 256;       // 16 lanes per edge
  const int gruBlocks  = (NN / 16 + 7) / 8;     // 8 waves per 256-thread block

  // One-time weight conversion to f16 (tiny; runs every call for determinism)
  cvt_f16_kernel<<<cvtBlocks, 256, 0, stream>>>(wih1, wih1h, nW);
  cvt_f16_kernel<<<cvtBlocks, 256, 0, stream>>>(whh1, whh1h, nW);
  cvt_f16_kernel<<<cvtBlocks, 256, 0, stream>>>(wih2, wih2h, nW);
  cvt_f16_kernel<<<cvtBlocks, 256, 0, stream>>>(whh2, whh2h, nW);

  // Layer 1
  zero_kernel<<<zeroBlocks, 256, 0, stream>>>((float4*)aggr, n4);
  scatter_kernel<<<scatBlocks, 256, 0, stream>>>(x, ei, ew, aggr);
  gru_kernel<<<gruBlocks, 256, 0, stream>>>(aggr, x, wih1h, whh1h, bih1, bhh1, h1);
  // Layer 2
  zero_kernel<<<zeroBlocks, 256, 0, stream>>>((float4*)aggr, n4);
  scatter_kernel<<<scatBlocks, 256, 0, stream>>>(h1, ei, ew, aggr);
  gru_kernel<<<gruBlocks, 256, 0, stream>>>(aggr, h1, wih2h, whh2h, bih2, bhh2, out);
}
